// Decoder_8375186227441
// MI455X (gfx1250) — matrix operand order repaired and checked by reading, not compiled, not run on hardware
//
#include <hip/hip_runtime.h>
#include <hip/hip_bf16.h>

#define BN   64
#define TENCN 512
#define TDECN 64
#define HD   1024
#define VVN  1024
#define AD   512

typedef __attribute__((ext_vector_type(16))) __bf16 v16bf;
typedef __attribute__((ext_vector_type(8)))  __bf16 v8bf;
typedef __attribute__((ext_vector_type(8)))  float  v8f;

__device__ __forceinline__ float fsigmoid(float x) { return 1.f / (1.f + __expf(-x)); }
__device__ __forceinline__ float ftanhf(float x)   { return 1.f - 2.f / (__expf(2.f * x) + 1.f); }

// ---------------------------------------------------------------------------
// WMMA GEMM: C[m,n] = sum_k A[m,k] * W[n,k] (+ bias[n])
// Each wave32 computes a 64-row x 16-col strip = 4 stacked 16x16 tiles, so the
// B (weight) fragment is loaded ONCE per K-step and reused by 4 WMMAs (the
// weight stream is the dominant traffic; activations are tiny and L2-hot).
// Fragment layouts (ISA 7.12.2):
//   A 16x32 bf16: lane l -> row m = l%16, half = l/16;
//     elem i<8 -> k = k0 + half*8 + i ; elem i>=8 -> k = k0 + 16 + half*8 + (i-8)
//   B 32x16 bf16: lane l -> col n = l%16 (row n of W); same k mapping as A
//   D f32 16x16:  lane l -> col n = l%16, VGPR i -> row m = i + 8*half
// ---------------------------------------------------------------------------
template<bool AF32, bool OBF16>
__global__ void k_gemm(const void* __restrict__ Aptr, int lda,
                       const __bf16* __restrict__ W, int ldw,
                       const float* __restrict__ bias,
                       void* __restrict__ Cptr, int ldc,
                       int M, int N, int K, int total_waves)
{
    int wave = (int)((blockIdx.x * blockDim.x + threadIdx.x) >> 5);
    if (wave >= total_waves) return;                 // wave-uniform: EXEC stays all-1s
    int lane = threadIdx.x & 31;
    int ntile = N >> 4;
    int mg = wave / ntile, nt = wave - mg * ntile;   // mg: 64-row group index
    int half = lane >> 4, r = lane & 15;
    int m0 = mg * 64;
    const float*  Af = (const float*)Aptr;
    const __bf16* Ab = (const __bf16*)Aptr;
    size_t arow0 = (size_t)(m0 +  0 + r) * (size_t)lda;
    size_t arow1 = (size_t)(m0 + 16 + r) * (size_t)lda;
    size_t arow2 = (size_t)(m0 + 32 + r) * (size_t)lda;
    size_t arow3 = (size_t)(m0 + 48 + r) * (size_t)lda;
    size_t wrow  = (size_t)(nt * 16 + r) * (size_t)ldw;
    v8f acc0 = {}, acc1 = {}, acc2 = {}, acc3 = {};

#define AFRAG_WMMA(AROW, ACC)                                                          \
    {                                                                                  \
        v16bf a;                                                                       \
        if (AF32) {                                                                    \
            const float* pa = Af + (AROW) + k0 + half * 8;                             \
            v8f x0 = *(const v8f*)pa;                                                  \
            v8f x1 = *(const v8f*)(pa + 16);                                           \
            _Pragma("unroll")                                                          \
            for (int i = 0; i < 8; ++i) { a[i] = (__bf16)x0[i]; a[i + 8] = (__bf16)x1[i]; } \
        } else {                                                                       \
            const __bf16* pa = Ab + (AROW) + k0 + half * 8;                            \
            v8bf x0 = *(const v8bf*)pa;                                                \
            v8bf x1 = *(const v8bf*)(pa + 16);                                         \
            _Pragma("unroll")                                                          \
            for (int i = 0; i < 8; ++i) { a[i] = x0[i]; a[i + 8] = x1[i]; }            \
        }                                                                              \
        ACC = __builtin_amdgcn_wmma_f32_16x16x32_bf16(false, a, false, b,              \
                                                      (short)0, ACC, false, false);    \
    }

    for (int k0 = 0; k0 < K; k0 += 32) {
        const __bf16* pw = W + wrow + k0 + half * 8;
        __builtin_prefetch((const void*)(pw + 256), 0, 1);   // global_prefetch_b8
        v8bf w0 = *(const v8bf*)pw, w1 = *(const v8bf*)(pw + 16);
        v16bf b = __builtin_shufflevector(w0, w1, 0,1,2,3,4,5,6,7,8,9,10,11,12,13,14,15); // ONE weight frag, 4 WMMAs
        AFRAG_WMMA(arow0, acc0)
        AFRAG_WMMA(arow1, acc1)
        AFRAG_WMMA(arow2, acc2)
        AFRAG_WMMA(arow3, acc3)
    }
#undef AFRAG_WMMA

    int n = nt * 16 + r;
    float bvv = bias ? bias[n] : 0.f;
    v8f accs[4] = { acc0, acc1, acc2, acc3 };
#pragma unroll
    for (int mi = 0; mi < 4; ++mi) {
#pragma unroll
        for (int i = 0; i < 8; ++i) {
            float v = accs[mi][i] + bvv;
            size_t off = (size_t)(m0 + mi * 16 + i + 8 * half) * (size_t)ldc + n;
            if (OBF16) ((__bf16*)Cptr)[off] = (__bf16)v;
            else       ((float*)Cptr)[off]  = v;
        }
    }
}

// --------------------------- prep / elementwise ----------------------------
__global__ void k_cvt(__bf16* dst, const float* src, int n) {
    int i = blockIdx.x * blockDim.x + threadIdx.x;
    if (i < n) dst[i] = (__bf16)src[i];
}

// dst[n][0:K] = a[n][:], dst[n][K:2K] = b[n][:], bf16; total = N*2K
__global__ void k_pack2(__bf16* dst, const float* a, const float* b, int K, int total) {
    int i = blockIdx.x * blockDim.x + threadIdx.x;
    if (i >= total) return;
    int row = i / (2 * K);
    int col = i - row * 2 * K;
    float v = (col < K) ? a[(size_t)row * K + col] : b[(size_t)row * K + col - K];
    dst[i] = (__bf16)v;
}

__global__ void k_addb(float* dst, const float* a, const float* b, int n) {
    int i = blockIdx.x * blockDim.x + threadIdx.x;
    if (i < n) dst[i] = a[i] + b[i];
}

__global__ void k_init(float* c0, float* c1, __bf16* xcat0, __bf16* xcat1, float* loss) {
    int i = blockIdx.x * blockDim.x + threadIdx.x;
    if (i < BN * HD) { c0[i] = 0.f; c1[i] = 0.f; }
    if (i < BN * 2048) { xcat0[i] = (__bf16)0.f; xcat1[i] = (__bf16)0.f; }
    if (i == 0) *loss = 0.f;
}

// xcat0[:, 0:512] = emb_table[prev_token]
__global__ void k_embed(const int* __restrict__ tgt, const float* __restrict__ emb,
                        __bf16* __restrict__ xcat0, int t) {
    int i = blockIdx.x * blockDim.x + threadIdx.x;   // BN*AD
    if (i >= BN * AD) return;
    int b = i >> 9, a = i & (AD - 1);
    int prev = (t == 0) ? 0 : tgt[b * TDECN + t - 1];
    xcat0[(size_t)b * 2048 + a] = (__bf16)emb[(size_t)prev * AD + a];
}

// LSTM cell epilogue: gates[b, {i,f,g,o}*1024], c in/out f32, h -> two bf16 sinks
__global__ void k_lstm(const float* __restrict__ gates, float* __restrict__ c,
                       __bf16* __restrict__ dstA, int ldA, int offA,
                       __bf16* __restrict__ dstB, int ldB, int offB) {
    int i = blockIdx.x * blockDim.x + threadIdx.x;   // BN*HD
    if (i >= BN * HD) return;
    int b = i >> 10, j = i & (HD - 1);
    const float* g = gates + (size_t)b * 4096;
    float ig = fsigmoid(g[j]);
    float fg = fsigmoid(g[1024 + j]);
    float gg = ftanhf(g[2048 + j]);
    float og = fsigmoid(g[3072 + j]);
    float cn = fg * c[i] + ig * gg;
    c[i] = cn;
    __bf16 hb = (__bf16)(og * ftanhf(cn));
    dstA[(size_t)b * ldA + offA + j] = hb;
    dstB[(size_t)b * ldB + offB + j] = hb;
}

// per-batch dot-product attention: scores -> softmax -> weighted V, bf16 out
__global__ void k_attn(const __bf16* __restrict__ kmat, const __bf16* __restrict__ vmat,
                       const float* __restrict__ q, __bf16* __restrict__ xcat0) {
    __shared__ float sq[AD];
    __shared__ float sw[TENCN];
    __shared__ float red[256];
    int b = blockIdx.x, tid = threadIdx.x;
    for (int a = tid; a < AD; a += 256) sq[a] = q[b * AD + a];
    __syncthreads();
    float lmax = -3.0e38f;
    for (int t = tid; t < TENCN; t += 256) {
        const __bf16* kr = kmat + ((size_t)b * TENCN + t) * AD;
        float s = 0.f;
        for (int a = 0; a < AD; a += 8) {
            v8bf kv = *(const v8bf*)(kr + a);
#pragma unroll
            for (int j = 0; j < 8; ++j) s += (float)kv[j] * sq[a + j];
        }
        sw[t] = s;
        lmax = fmaxf(lmax, s);
    }
    red[tid] = lmax; __syncthreads();
    for (int o = 128; o > 0; o >>= 1) { if (tid < o) red[tid] = fmaxf(red[tid], red[tid + o]); __syncthreads(); }
    float mx = red[0]; __syncthreads();
    float lsum = 0.f;
    for (int t = tid; t < TENCN; t += 256) { float e = __expf(sw[t] - mx); sw[t] = e; lsum += e; }
    red[tid] = lsum; __syncthreads();
    for (int o = 128; o > 0; o >>= 1) { if (tid < o) red[tid] += red[tid + o]; __syncthreads(); }
    float inv = 1.f / red[0]; __syncthreads();
    for (int a = tid; a < AD; a += 256) {
        float acc = 0.f;
        for (int t = 0; t < TENCN; ++t)
            acc += sw[t] * (float)vmat[((size_t)b * TENCN + t) * AD + a];
        xcat0[(size_t)b * 2048 + AD + a] = (__bf16)(acc * inv);
    }
}

// LayerNorm + ReLU over 2048, f32 in, bf16 out
__global__ void k_ln(const float* __restrict__ x, const float* __restrict__ g,
                     const float* __restrict__ bb, __bf16* __restrict__ out) {
    __shared__ float red[256];
    int b = blockIdx.x, tid = threadIdx.x;
    const float* xr = x + (size_t)b * 2048;
    float s = 0.f;
    for (int j = tid; j < 2048; j += 256) s += xr[j];
    red[tid] = s; __syncthreads();
    for (int o = 128; o > 0; o >>= 1) { if (tid < o) red[tid] += red[tid + o]; __syncthreads(); }
    float mu = red[0] * (1.f / 2048.f); __syncthreads();
    float vs = 0.f;
    for (int j = tid; j < 2048; j += 256) { float d = xr[j] - mu; vs += d * d; }
    red[tid] = vs; __syncthreads();
    for (int o = 128; o > 0; o >>= 1) { if (tid < o) red[tid] += red[tid + o]; __syncthreads(); }
    float inv = rsqrtf(red[0] * (1.f / 2048.f) + 1e-5f); __syncthreads();
    for (int j = tid; j < 2048; j += 256) {
        float y = (xr[j] - mu) * inv * g[j] + bb[j];
        out[(size_t)b * 2048 + j] = (__bf16)fmaxf(y, 0.f);
    }
}

// mean cross-entropy over all (b,t): one wave per row of 1024 logits
__global__ void k_loss(const float* __restrict__ logits, const int* __restrict__ tgt,
                       float* __restrict__ loss) {
    int wave = (int)((blockIdx.x * blockDim.x + threadIdx.x) >> 5);
    int lane = threadIdx.x & 31;
    if (wave >= BN * TDECN) return;
    const float* row = logits + (size_t)wave * VVN;
    float mx = -3.0e38f;
    for (int j = lane; j < VVN; j += 32) mx = fmaxf(mx, row[j]);
    for (int o = 16; o > 0; o >>= 1) mx = fmaxf(mx, __shfl_xor(mx, o, 32));
    float s = 0.f;
    for (int j = lane; j < VVN; j += 32) s += __expf(row[j] - mx);
    for (int o = 16; o > 0; o >>= 1) s += __shfl_xor(s, o, 32);
    if (lane == 0) {
        float nll = (mx + __logf(s)) - row[tgt[wave]];
        atomicAdd(loss, nll * (1.f / (float)(BN * TDECN)));
    }
}

// ---------------------------------------------------------------------------
extern "C" void kernel_launch(void* const* d_in, const int* in_sizes, int n_in,
                              void* d_out, int out_size, void* d_ws, size_t ws_size,
                              hipStream_t stream)
{
    const float* enc  = (const float*)d_in[0];
    const int*   tgt  = (const int*)  d_in[1];
    const float* emb  = (const float*)d_in[2];
    const float* wq   = (const float*)d_in[3];
    const float* bq   = (const float*)d_in[4];
    const float* wk   = (const float*)d_in[5];
    const float* bk   = (const float*)d_in[6];
    const float* wv   = (const float*)d_in[7];
    const float* bv   = (const float*)d_in[8];
    const float* wih0 = (const float*)d_in[9];
    const float* whh0 = (const float*)d_in[10];
    const float* bih0 = (const float*)d_in[11];
    const float* bhh0 = (const float*)d_in[12];
    const float* wih1 = (const float*)d_in[13];
    const float* whh1 = (const float*)d_in[14];
    const float* bih1 = (const float*)d_in[15];
    const float* bhh1 = (const float*)d_in[16];
    const float* wm1  = (const float*)d_in[17];
    const float* bm1  = (const float*)d_in[18];
    const float* lng  = (const float*)d_in[19];
    const float* lnb  = (const float*)d_in[20];
    const float* wm2  = (const float*)d_in[21];
    const float* bm2  = (const float*)d_in[22];

    float* logits = (float*)d_out;
    float* loss   = logits + (size_t)BN * TDECN * VVN;

    // ---- workspace carve (all 256B aligned) ----
    char* p = (char*)d_ws;
    auto carve = [&](size_t bytes) -> void* {
        void* r = (void*)p; p += (bytes + 255) & ~(size_t)255; return r;
    };
    __bf16* kbf    = (__bf16*)carve((size_t)BN * TENCN * AD * 2);   // 33.5 MB
    __bf16* vbf    = (__bf16*)carve((size_t)BN * TENCN * AD * 2);   // 33.5 MB
    __bf16* wcat0  = (__bf16*)carve((size_t)4096 * 2048 * 2);       // 16.8 MB
    __bf16* wcat1  = (__bf16*)carve((size_t)4096 * 2048 * 2);       // 16.8 MB
    __bf16* wqbf   = (__bf16*)carve((size_t)AD * HD * 2);
    __bf16* wkbf   = (__bf16*)carve((size_t)AD * HD * 2);
    __bf16* wvbf   = (__bf16*)carve((size_t)AD * HD * 2);
    __bf16* wm1bf  = (__bf16*)carve((size_t)2048 * 1024 * 2);
    __bf16* wm2bf  = (__bf16*)carve((size_t)1024 * 2048 * 2);
    float*  b0     = (float*)carve(4096 * 4);
    float*  b1     = (float*)carve(4096 * 4);
    __bf16* xcat0  = (__bf16*)carve((size_t)BN * 2048 * 2);  // [e | attn | h0]
    __bf16* xcat1  = (__bf16*)carve((size_t)BN * 2048 * 2);  // [h0n | h1]
    __bf16* h1bf   = (__bf16*)carve((size_t)BN * 1024 * 2);
    __bf16* xmlpbf = (__bf16*)carve((size_t)BN * 2048 * 2);
    float*  gates  = (float*)carve((size_t)BN * 4096 * 4);
    float*  c0     = (float*)carve((size_t)BN * 1024 * 4);
    float*  c1     = (float*)carve((size_t)BN * 1024 * 4);
    float*  qf     = (float*)carve((size_t)BN * 512 * 4);
    float*  xmlpf  = (float*)carve((size_t)BN * 2048 * 4);

    // ---- weight prep (bf16 conversion / K-concat packing, bias folding) ----
    k_pack2<<<32768, 256, 0, stream>>>(wcat0, wih0, whh0, 1024, 4096 * 2048);
    k_pack2<<<32768, 256, 0, stream>>>(wcat1, wih1, whh1, 1024, 4096 * 2048);
    k_cvt<<<2048, 256, 0, stream>>>(wqbf, wq, AD * HD);
    k_cvt<<<2048, 256, 0, stream>>>(wkbf, wk, AD * HD);
    k_cvt<<<2048, 256, 0, stream>>>(wvbf, wv, AD * HD);
    k_cvt<<<8192, 256, 0, stream>>>(wm1bf, wm1, 2048 * 1024);
    k_cvt<<<8192, 256, 0, stream>>>(wm2bf, wm2, 1024 * 2048);
    k_addb<<<16, 256, 0, stream>>>(b0, bih0, bhh0, 4096);
    k_addb<<<16, 256, 0, stream>>>(b1, bih1, bhh1, 4096);
    k_init<<<512, 256, 0, stream>>>(c0, c1, xcat0, xcat1, loss);

    // ---- K/V precompute: [B*TENC,H] fp32 @ wk/wvᵀ -> bf16 [B*TENC,A] ----
    {
        int waves = (BN * TENCN / 64) * (AD / 16);          // 16384
        k_gemm<true, true><<<waves / 8, 256, 0, stream>>>(enc, HD, wkbf, HD, bk,
                                                          kbf, AD, BN * TENCN, AD, HD, waves);
        k_gemm<true, true><<<waves / 8, 256, 0, stream>>>(enc, HD, wvbf, HD, bv,
                                                          vbf, AD, BN * TENCN, AD, HD, waves);
    }

    // ---- 64 sequential decoder steps ----
    for (int t = 0; t < TDECN; ++t) {
        k_embed<<<128, 256, 0, stream>>>(tgt, emb, xcat0, t);
        // LSTM0 gates: [ci|h0] (K=2048) @ [w_ih0|w_hh0]ᵀ -> f32 [64,4096]
        k_gemm<false, false><<<32, 256, 0, stream>>>(xcat0, 2048, wcat0, 2048, b0,
                                                     gates, 4096, BN, 4096, 2048, 256);
        k_lstm<<<256, 256, 0, stream>>>(gates, c0, xcat1, 2048, 0, xcat0, 2048, 1024);
        // LSTM1 gates: [h0n|h1] @ [w_ih1|w_hh1]ᵀ
        k_gemm<false, false><<<32, 256, 0, stream>>>(xcat1, 2048, wcat1, 2048, b1,
                                                     gates, 4096, BN, 4096, 2048, 256);
        k_lstm<<<256, 256, 0, stream>>>(gates, c1, xcat1, 2048, 1024, h1bf, 1024, 0);
        // q = h1n @ wqᵀ + bq
        k_gemm<false, false><<<4, 256, 0, stream>>>(h1bf, 1024, wqbf, 1024, bq,
                                                    qf, 512, BN, 512, 1024, 32);
        // attention -> xcat0[:, 512:1024] (consumed by next step's LSTM0)
        k_attn<<<BN, 256, 0, stream>>>(kbf, vbf, qf, xcat0);
        // MLP head
        k_gemm<false, false><<<16, 256, 0, stream>>>(h1bf, 1024, wm1bf, 1024, bm1,
                                                     xmlpf, 2048, BN, 2048, 1024, 128);
        k_ln<<<BN, 256, 0, stream>>>(xmlpf, lng, lnb, xmlpbf);
        // logits_t -> d_out[b, t, :]  (ldc = TDEC*V strides the b dimension)
        k_gemm<false, false><<<8, 256, 0, stream>>>(xmlpbf, 2048, wm2bf, 2048, bm2,
                                                    logits + (size_t)t * VVN, TDECN * VVN,
                                                    BN, VVN, 2048, 64);
    }

    // ---- mean cross-entropy loss over all (b,t) ----
    k_loss<<<512, 256, 0, stream>>>(logits, tgt, loss);
}
